// PNA_concave_26843545600707
// MI455X (gfx1250) — matrix operand order repaired and checked by reading, not compiled, not run on hardware
//
#include <hip/hip_runtime.h>

#define NN 10000          // nodes
#define NE 160000         // edges
#define NG 64             // graphs
#define FDIM 80           // F
#define TDIM 5            // towers
#define NCOL 400          // T*F
#define KDIM 160          // 2F
#define FOUT 16
#define NLAYERS 4
#define NPB 8             // nodes per block in k_post

typedef __attribute__((ext_vector_type(16))) _Float16 v16h;
typedef __attribute__((ext_vector_type(8)))  float    v8f;

// order-preserving float<->uint encoding for atomic min/max
__device__ __forceinline__ unsigned fenc(float f) {
  unsigned u = __float_as_uint(f);
  return ((int)u >= 0) ? (u | 0x80000000u) : ~u;
}
__device__ __forceinline__ float fdec(unsigned u) {
  return __uint_as_float((u & 0x80000000u) ? (u & 0x7FFFFFFFu) : ~u);
}

__global__ void k_fill_f(float* p, float v, int n) {
  for (int i = blockIdx.x*blockDim.x + threadIdx.x; i < n; i += gridDim.x*blockDim.x) p[i] = v;
}

// interleaved aggregation buffer: per (node, col) -> {sum.f32, sumsq.f32, min.enc, max.enc}
__global__ void k_reset_agg(uint4* a, int n) {
  uint4 init = make_uint4(0u, 0u, 0xFFFFFFFFu, 0u);
  for (int i = blockIdx.x*blockDim.x + threadIdx.x; i < n; i += gridDim.x*blockDim.x)
    a[i] = init;
}

__global__ void k_deg(const int* __restrict__ dst, float* __restrict__ deg) {
  int e = blockIdx.x*blockDim.x + threadIdx.x;
  if (e < NE) atomicAdd(&deg[dst[e]], 1.0f);
}

__global__ void k_avg_log_deg(const float* __restrict__ deg, float* out) {
  __shared__ float sh[256];
  float s = 0.f;
  for (int i = threadIdx.x; i < NN; i += 256) s += logf(deg[i] + 1.0f);
  sh[threadIdx.x] = s; __syncthreads();
  for (int o = 128; o > 0; o >>= 1) {
    if (threadIdx.x < o) sh[threadIdx.x] += sh[threadIdx.x + o];
    __syncthreads();
  }
  if (threadIdx.x == 0) out[0] = sh[0] / (float)NN;
}

__global__ void k_init_h(const float* __restrict__ x, const float* __restrict__ W,
                         const float* __restrict__ b, float* __restrict__ h,
                         _Float16* __restrict__ h16) {
  int i = blockIdx.x*blockDim.x + threadIdx.x;
  if (i >= NN*FDIM) return;
  int n = i / FDIM, f = i % FDIM;
  float v = x[n*2 + 0] * W[f] + b[f];
  h[i] = v; h16[i] = (_Float16)v;
}

// preW[l]: [T][160][80] f32  ->  preWT: [400][160] f16 (row c = t*80+o)
__global__ void k_convW(const float* __restrict__ preW, _Float16* __restrict__ preWT) {
  int i = blockIdx.x*blockDim.x + threadIdx.x;
  if (i >= NCOL*KDIM) return;
  int c = i / KDIM, k = i % KDIM;
  int t = c / FDIM, o = c % FDIM;
  preWT[i] = (_Float16)preW[((size_t)t*KDIM + k)*FDIM + o];
}

// ---- core: 16 edges per block, WMMA f16 GEMM (16x400x160) + atomic scatter ----
__global__ __launch_bounds__(128) void k_edge_gemm(
    const _Float16* __restrict__ h16, const _Float16* __restrict__ preWT,
    const float* __restrict__ prebL, const int* __restrict__ srcI,
    const int* __restrict__ dstI, uint4* __restrict__ agg)
{
  __shared__ _Float16 As[16 * KDIM];
  __shared__ int dnode[16];
  const int e0 = blockIdx.x * 16;
  const int tid = threadIdx.x;
  if (tid < 16) dnode[tid] = dstI[e0 + tid];
  // stage A: row m = edge, cols 0..79 = h[dst], 80..159 = h[src]
  for (int i = tid; i < 16*40; i += 128) {
    int m = i / 40, ch = i % 40;
    int node = (ch < 20) ? dstI[e0 + m] : srcI[e0 + m];
    int off = (ch % 20) * 4;
    uint2 v = *(const uint2*)(h16 + (size_t)node*FDIM + off);
    *(uint2*)(As + m*KDIM + (ch < 20 ? off : FDIM + off)) = v;
  }
  __syncthreads();

  const int lane = tid & 31, wave = tid >> 5;
  const int half = lane >> 4, mrow = lane & 15;

  // A fragments in contiguous k order (k = 16*half + s, same as B) (held for all 5 K-chunks)
  union AV { v16h v; unsigned u[8]; };
  AV a[5];
  #pragma unroll
  for (int kc = 0; kc < 5; ++kc) {
    #pragma unroll
    for (int i = 0; i < 8; ++i) {
      int k0 = kc*32 + 16*half + 2*i;
      a[kc].u[i] = *(const unsigned*)(As + mrow*KDIM + k0);
    }
  }

  union BV { v16h v; uint4 q[2]; };
  for (int nt = wave; nt < 25; nt += 4) {
    const int n = nt*16 + mrow;                 // output column (t*80+o)
    const float bias = prebL[n];
    // prefetch ALL B fragments for this tile (10x b128 in one clause, one wait)
    const _Float16* bp = preWT + (size_t)n*KDIM + half*16;
    BV b[5];
    #pragma unroll
    for (int kc = 0; kc < 5; ++kc) {
      b[kc].q[0] = *(const uint4*)(bp + kc*32);
      b[kc].q[1] = *(const uint4*)(bp + kc*32 + 8);
    }
    v8f acc = {bias,bias,bias,bias,bias,bias,bias,bias};
    #pragma unroll
    for (int kc = 0; kc < 5; ++kc)
      acc = __builtin_amdgcn_wmma_f32_16x16x32_f16(false, a[kc].v, false, b[kc].v,
                                                   (short)0, acc, false, false);
    #pragma unroll
    for (int r = 0; r < 8; ++r) {
      int m = r + 8*half;
      int d = dnode[m];
      float v = acc[r];
      // single base address; 4 atomics at immediate offsets, same cacheline
      unsigned* p = (unsigned*)(agg + ((size_t)d*NCOL + n));
      atomicAdd((float*)p,     v);
      atomicAdd((float*)p + 1, v*v);
      unsigned e = fenc(v);
      atomicMin(p + 2, e);
      atomicMax(p + 3, e);
    }
  }
}

// per-node post transform, NPB nodes per block (amortize postW reads)
__global__ __launch_bounds__(256) void k_post(
    const float* __restrict__ h, const uint4* __restrict__ agg,
    const float* __restrict__ deg, const float* __restrict__ avgld,
    const float* __restrict__ postWL, const float* __restrict__ postbL,
    const float* __restrict__ linWL, const float* __restrict__ linbL,
    float* __restrict__ y)
{
  __shared__ float sMean[NPB*NCOL], sMin[NPB*NCOL], sMax[NPB*NCOL], sStd[NPB*NCOL];
  __shared__ float sH[NPB*FDIM];
  __shared__ float sOut[NPB*TDIM*FOUT];
  __shared__ float sCnt[NPB], sHas[NPB], sAmp[NPB], sAtt[NPB];
  const int node0 = blockIdx.x * NPB;
  const int tid = threadIdx.x;
  if (tid < NPB) {
    float dg = deg[node0 + tid];
    float cnt = fmaxf(dg, 1.0f);
    float logd = logf(cnt + 1.0f);
    float av = avgld[0];
    sCnt[tid] = cnt;
    sHas[tid] = (dg > 0.0f) ? 1.0f : 0.0f;
    sAmp[tid] = logd / av;
    sAtt[tid] = av / logd;
  }
  __syncthreads();
  for (int idx = tid; idx < NPB*NCOL; idx += 256) {
    int nr = idx / NCOL, c = idx % NCOL;
    uint4 q = agg[(size_t)(node0 + nr)*NCOL + c];
    float cnt = sCnt[nr];
    float has = sHas[nr];
    float me = __uint_as_float(q.x) / cnt;
    float ms = __uint_as_float(q.y) / cnt;
    sMean[idx] = me;
    sMin[idx] = has * fdec(q.z);
    sMax[idx] = has * fdec(q.w);
    sStd[idx] = sqrtf(fmaxf(ms - me*me, 0.0f) + 1e-5f);
  }
  for (int idx = tid; idx < NPB*FDIM; idx += 256)
    sH[idx] = h[(size_t)node0*FDIM + idx];
  __syncthreads();
  // out[nr][t][o] = postb + [h | agg | amp | att] . postW column  (amp/att folded)
  for (int it = tid; it < NPB*TDIM*FOUT; it += 256) {
    int nr = it / (TDIM*FOUT);
    int to = it % (TDIM*FOUT);
    int t = to / FOUT, o = to % FOUT;
    const float* W = postWL + (size_t)t * (13*FDIM) * FOUT;
    const float s_amp = sAmp[nr], s_att = sAtt[nr];
    float acc = postbL[to];
    const float* hv = sH + nr*FDIM;
    for (int f = 0; f < FDIM; ++f) acc += hv[f] * W[f*FOUT + o];
    const float* G[4] = {sMean, sMin, sMax, sStd};
    #pragma unroll
    for (int g = 0; g < 4; ++g) {
      const float* Wg = W + (size_t)(FDIM + g*FDIM)*FOUT;
      const float* Wa = W + (size_t)(5*FDIM + g*FDIM)*FOUT;
      const float* Wt = W + (size_t)(9*FDIM + g*FDIM)*FOUT;
      const float* gv = G[g] + nr*NCOL + t*FDIM;
      for (int f = 0; f < FDIM; ++f) {
        float w = Wg[f*FOUT+o] + s_amp*Wa[f*FOUT+o] + s_att*Wt[f*FOUT+o];
        acc += gv[f] * w;
      }
    }
    sOut[it] = acc;
  }
  __syncthreads();
  for (int it = tid; it < NPB*FDIM; it += 256) {
    int nr = it / FDIM, oc = it % FDIM;
    float acc = linbL[oc];
    const float* ov = sOut + nr*TDIM*FOUT;
    for (int k = 0; k < TDIM*FOUT; ++k) acc += ov[k] * linWL[k*FDIM + oc];
    y[(size_t)(node0 + nr)*FDIM + oc] = acc;
  }
}

__global__ void k_bn_stats(const float* __restrict__ y, float* mu, float* inv) {
  __shared__ float s1[256], s2[256];
  const int f = blockIdx.x;
  float a = 0.f, b = 0.f;
  for (int n = threadIdx.x; n < NN; n += 256) {
    float v = y[(size_t)n*FDIM + f];
    a += v; b += v*v;
  }
  s1[threadIdx.x] = a; s2[threadIdx.x] = b; __syncthreads();
  for (int o = 128; o > 0; o >>= 1) {
    if (threadIdx.x < o) { s1[threadIdx.x] += s1[threadIdx.x+o]; s2[threadIdx.x] += s2[threadIdx.x+o]; }
    __syncthreads();
  }
  if (threadIdx.x == 0) {
    float m = s1[0] / (float)NN;
    float var = s2[0] / (float)NN - m*m;
    mu[f] = m; inv[f] = rsqrtf(fmaxf(var, 0.f) + 1e-5f);
  }
}

__global__ void k_bn_apply(const float* __restrict__ y, const float* __restrict__ mu,
                           const float* __restrict__ inv, const float* __restrict__ gm,
                           const float* __restrict__ bt, float* __restrict__ h,
                           _Float16* __restrict__ h16) {
  int i = blockIdx.x*blockDim.x + threadIdx.x;
  if (i >= NN*FDIM) return;
  int f = i % FDIM;
  float v = (y[i] - mu[f]) * inv[f] * gm[f] + bt[f];
  v = fmaxf(v, 0.f);
  h[i] = v; h16[i] = (_Float16)v;
}

__global__ void k_final_prep(const float* __restrict__ h, const float* __restrict__ x,
                             float* __restrict__ lc, float* __restrict__ ls) {
  int i = blockIdx.x*blockDim.x + threadIdx.x;
  if (i >= NN*40) return;
  int n = i / 40, f = i % 40;
  float xc = h[(size_t)n*FDIM + f] * x[n*2 + 1] + h[(size_t)n*FDIM + 40 + f];
  lc[i] = logf(xc + 1e-6f);
  ls[i] = 0.f;
}

__global__ void k_edge_log(const int* __restrict__ srcI, const int* __restrict__ dstI,
                           const float* __restrict__ lc, float* __restrict__ ls) {
  int i = blockIdx.x*blockDim.x + threadIdx.x;
  if (i >= NE*40) return;
  int e = i / 40, f = i % 40;
  atomicAdd(&ls[(size_t)dstI[e]*40 + f], lc[(size_t)srcI[e]*40 + f]);
}

__global__ void k_pool(const int* __restrict__ batch, const float* __restrict__ lc,
                       const float* __restrict__ ls, float* __restrict__ pooled,
                       float* __restrict__ gcnt) {
  int i = blockIdx.x*blockDim.x + threadIdx.x;
  if (i >= NN*40) return;
  int n = i / 40, f = i % 40;
  float v = expf(ls[i] + lc[i]);
  int g = batch[n];
  atomicAdd(&pooled[g*40 + f], v);
  if (f == 0) atomicAdd(&gcnt[g], 1.f);
}

__global__ void k_head(const float* __restrict__ pooled, const float* __restrict__ gcnt,
                       const float* __restrict__ mlW, const float* __restrict__ mlb,
                       const float* __restrict__ m1W, const float* __restrict__ m1b,
                       const float* __restrict__ m2W, const float* __restrict__ m2b,
                       float* __restrict__ out) {
  int g = threadIdx.x;
  if (g >= NG) return;
  float p[40];
  float c = fmaxf(gcnt[g], 1.f);
  for (int i = 0; i < 40; ++i) p[i] = pooled[g*40 + i] / c;
  float xlin = mlb[0];
  for (int i = 0; i < 40; ++i) xlin += p[i] * mlW[i];
  float xcv = m2b[0];
  for (int j = 0; j < 20; ++j) {
    float t = m1b[j];
    for (int i = 0; i < 40; ++i) t += p[i] * m1W[i*20 + j];
    t = 20.f - fmaxf(t, 0.f);
    xcv += t * m2W[j];
  }
  out[g] = xcv + xlin;
}

extern "C" void kernel_launch(void* const* d_in, const int* in_sizes, int n_in,
                              void* d_out, int out_size, void* d_ws, size_t ws_size,
                              hipStream_t stream) {
  const float* x        = (const float*)d_in[0];
  const int*   ei       = (const int*)d_in[1];
  const int*   srcI     = ei;
  const int*   dstI     = ei + NE;
  const int*   batch    = (const int*)d_in[2];
  const float* preLinW  = (const float*)d_in[3];
  const float* preLinB  = (const float*)d_in[4];
  const float* preW     = (const float*)d_in[5];
  const float* preB     = (const float*)d_in[6];
  const float* postW    = (const float*)d_in[7];
  const float* postB    = (const float*)d_in[8];
  const float* linW     = (const float*)d_in[9];
  const float* linB     = (const float*)d_in[10];
  const float* bnG      = (const float*)d_in[11];
  const float* bnB      = (const float*)d_in[12];
  const float* mlW      = (const float*)d_in[13];
  const float* mlb      = (const float*)d_in[14];
  const float* m1W      = (const float*)d_in[15];
  const float* m1b      = (const float*)d_in[16];
  const float* m2W      = (const float*)d_in[17];
  const float* m2b      = (const float*)d_in[18];
  float* outp = (float*)d_out;

  // workspace carve-up (~76 MB total)
  char* ws = (char*)d_ws;
  size_t off = 0;
  auto carve = [&](size_t bytes) -> char* {
    char* p = ws + off;
    off = (off + bytes + 255) & ~(size_t)255;
    return p;
  };
  float*     deg    = (float*)carve(NN*4);
  float*     avgld  = (float*)carve(64);
  float*     h      = (float*)carve((size_t)NN*FDIM*4);
  _Float16*  h16    = (_Float16*)carve((size_t)NN*FDIM*2);
  _Float16*  preWT  = (_Float16*)carve((size_t)NCOL*KDIM*2);
  uint4*     agg    = (uint4*)carve((size_t)NN*NCOL*16);
  float*     y      = (float*)carve((size_t)NN*FDIM*4);
  float*     mu     = (float*)carve(FDIM*4);
  float*     inv    = (float*)carve(FDIM*4);
  float*     lc     = (float*)carve((size_t)NN*40*4);
  float*     ls     = (float*)carve((size_t)NN*40*4);
  float*     pooled = (float*)carve(NG*40*4);
  float*     gcnt   = (float*)carve(NG*4);

  k_fill_f<<<40, 256, 0, stream>>>(deg, 0.f, NN);
  k_deg<<<(NE + 255)/256, 256, 0, stream>>>(dstI, deg);
  k_avg_log_deg<<<1, 256, 0, stream>>>(deg, avgld);
  k_init_h<<<(NN*FDIM + 255)/256, 256, 0, stream>>>(x, preLinW, preLinB, h, h16);

  for (int l = 0; l < NLAYERS; ++l) {
    k_reset_agg<<<4096, 256, 0, stream>>>(agg, NN*NCOL);
    k_convW<<<(NCOL*KDIM + 255)/256, 256, 0, stream>>>(preW + (size_t)l*TDIM*KDIM*FDIM, preWT);
    k_edge_gemm<<<NE/16, 128, 0, stream>>>(h16, preWT, preB + (size_t)l*NCOL,
                                           srcI, dstI, agg);
    k_post<<<NN/NPB, 256, 0, stream>>>(h, agg, deg, avgld,
                                       postW + (size_t)l*TDIM*13*FDIM*FOUT,
                                       postB + (size_t)l*TDIM*FOUT,
                                       linW + (size_t)l*FDIM*FDIM,
                                       linB + (size_t)l*FDIM, y);
    k_bn_stats<<<FDIM, 256, 0, stream>>>(y, mu, inv);
    k_bn_apply<<<(NN*FDIM + 255)/256, 256, 0, stream>>>(y, mu, inv, bnG + l*FDIM,
                                                        bnB + l*FDIM, h, h16);
  }

  k_final_prep<<<(NN*40 + 255)/256, 256, 0, stream>>>(h, x, lc, ls);
  k_edge_log<<<(NE*40 + 255)/256, 256, 0, stream>>>(srcI, dstI, lc, ls);
  k_fill_f<<<1, 256, 0, stream>>>(pooled, 0.f, NG*40);
  k_fill_f<<<1, 64, 0, stream>>>(gcnt, 0.f, NG);
  k_pool<<<(NN*40 + 255)/256, 256, 0, stream>>>(batch, lc, ls, pooled, gcnt);
  k_head<<<1, 64, 0, stream>>>(pooled, gcnt, mlW, mlb, m1W, m1b, m2W, m2b, outp);
}
